// MHSA_16312285790842
// MI455X (gfx1250) — matrix operand order repaired and checked by reading, not compiled, not run on hardware
//
#include <hip/hip_runtime.h>
#include <hip/hip_bf16.h>

// ---------------------------------------------------------------------------
// MHSA (Swin-window style) for MI455X / gfx1250, wave32 + WMMA f16 +
// async global->LDS double-buffered GEMM staging (ASYNCcnt pipeline).
//   B=64, N=256, C=768, H=12, Dh=64
// ---------------------------------------------------------------------------

typedef __attribute__((ext_vector_type(16))) _Float16 v16h;
typedef __attribute__((ext_vector_type(8)))  float    v8f;

#define PLANES 768
#define HEADS  12
#define NSEQ   256
#define BATCH  64
#define DH     64
#define SCALE_Q 0.125f            // 64^-0.5
#define MROWS  (BATCH * NSEQ)     // 16384
#define C3     (3 * PLANES)       // 2304

union HalfVec { v16h h; uint4 u[2]; };

// gfx1250 async DMA: global -> LDS, 16B per lane, tracked by ASYNCcnt.
//   global_load_async_to_lds_b128 vdst(lds byte addr), vaddr(u32 off), saddr
__device__ __forceinline__ void async_ld_b128(unsigned lds_off, unsigned voff,
                                              const void* base)
{
    asm volatile("global_load_async_to_lds_b128 %0, %1, %2"
                 :: "v"(lds_off), "v"(voff),
                    "s"((unsigned long long)(uintptr_t)base)
                 : "memory");
}
__device__ __forceinline__ unsigned lds_off_of(const void* p)
{
    return (unsigned)(uintptr_t)p;   // LDS aperture: offset = addr[31:0]
}

// ===========================================================================
// f16 WMMA GEMM:  C(MxN) = A(MxK,f16,row) * BT(NxK,f16,row)^T + bias
// block = 128 threads (4 waves), block tile 128x128, wave tile 64x64,
// K stepped by 32 with double-buffered async global->LDS staging.
// LDS tile rows padded to 40 halves (80 B) -> conflict-free ds_load_b128.
// ===========================================================================
#define TROW 40   // padded LDS row stride in halves

template<int OUT_F16>
__global__ __launch_bounds__(128) void gemm_wmma_kernel(
    const _Float16* __restrict__ A, const _Float16* __restrict__ BT,
    const float* __restrict__ bias, void* __restrict__ Cout,
    int M, int N, int K)
{
    __shared__ _Float16 smA[2][128 * TROW];
    __shared__ _Float16 smB[2][128 * TROW];

    const int tid    = threadIdx.x;
    const int lane   = tid & 31;
    const int wave   = tid >> 5;
    const int lane16 = lane & 15;
    const int laneHi = lane >> 4;
    const int m_block = blockIdx.x * 128;
    const int n_block = blockIdx.y * 128;
    const int waveM  = wave >> 1;
    const int waveN  = wave & 1;

    // stage one 128x32-half tile of A and of B into LDS buffer `buf`
    // 512 chunks of 16B per matrix; thread tid copies chunks i*128+tid
    // (consecutive tids -> consecutive 16B segments: coalesced)
    auto stage = [&](int buf, int k0) {
        #pragma unroll
        for (int i = 0; i < 4; ++i) {
            const int c   = i * 128 + tid;
            const int row = c >> 2;          // 0..127
            const int seg = c & 3;           // 16B segment within row
            const unsigned ga =
                (unsigned)(((unsigned)(m_block + row) * (unsigned)K + k0 + seg * 8) * 2);
            async_ld_b128(lds_off_of(&smA[buf][row * TROW + seg * 8]), ga, A);
            const unsigned gb =
                (unsigned)(((unsigned)(n_block + row) * (unsigned)K + k0 + seg * 8) * 2);
            async_ld_b128(lds_off_of(&smB[buf][row * TROW + seg * 8]), gb, BT);
        }
    };

    v8f acc[4][4];
    #pragma unroll
    for (int i = 0; i < 4; ++i)
        #pragma unroll
        for (int j = 0; j < 4; ++j)
            acc[i][j] = {};

    const int KSTEPS = K / 32;
    stage(0, 0);                               // prologue: 8 async ops in flight

    for (int kk = 0; kk < KSTEPS; ++kk) {
        if (kk + 1 < KSTEPS) {
            stage((kk + 1) & 1, (kk + 1) * 32);          // 8 more in flight
            asm volatile("s_wait_asynccnt 0x8" ::: "memory");  // stage kk done
        } else {
            asm volatile("s_wait_asynccnt 0x0" ::: "memory");
        }
        __syncthreads();

        const _Float16* sa = smA[kk & 1];
        const _Float16* sb = smB[kk & 1];

        v16h af[4], bf[4];
        #pragma unroll
        for (int i = 0; i < 4; ++i) {
            // A 16x32 frag: halves 0-7 -> K=laneHi*8.., halves 8-15 -> +16
            const _Float16* p = sa + (waveM * 64 + i * 16 + lane16) * TROW;
            HalfVec r;
            r.u[0] = *(const uint4*)(p + laneHi * 8);
            r.u[1] = *(const uint4*)(p + 16 + laneHi * 8);
            af[i] = r.h;
        }
        #pragma unroll
        for (int j = 0; j < 4; ++j) {
            // B 32x16 frag: halves 0-7 -> K=laneHi*8.., halves 8-15 -> +16
            const _Float16* p = sb + (waveN * 64 + j * 16 + lane16) * TROW + laneHi * 8;
            HalfVec r;
            r.u[0] = *(const uint4*)(p);
            r.u[1] = *(const uint4*)(p + 16);
            bf[j] = r.h;
        }
        #pragma unroll
        for (int i = 0; i < 4; ++i)
            #pragma unroll
            for (int j = 0; j < 4; ++j)
                acc[i][j] = __builtin_amdgcn_wmma_f32_16x16x32_f16(
                    false, af[i], false, bf[j], (short)0, acc[i][j], false, false);

        __syncthreads();   // before next stage overwrites the other buffer
    }

    // Epilogue: C VGPR r -> row r + laneHi*8, lane16 -> col
    #pragma unroll
    for (int j = 0; j < 4; ++j) {
        const int col = n_block + waveN * 64 + j * 16 + lane16;
        const float bc = bias[col];
        #pragma unroll
        for (int i = 0; i < 4; ++i) {
            #pragma unroll
            for (int r = 0; r < 8; ++r) {
                const int row = m_block + waveM * 64 + i * 16 + laneHi * 8 + r;
                const float v = acc[i][j][r] + bc;
                const size_t idx = (size_t)row * N + col;
                if (OUT_F16) ((_Float16*)Cout)[idx] = (_Float16)v;
                else         ((float*)Cout)[idx]    = v;
            }
        }
    }
}

// ===========================================================================
// Prep kernels
// ===========================================================================
__global__ __launch_bounds__(256) void f32_to_f16_kernel(
    const float* __restrict__ src, _Float16* __restrict__ dst, int n)
{
    int id = blockIdx.x * 256 + threadIdx.x;
    if (id < n) dst[id] = (_Float16)src[id];
}

// src: (K,N) f32  ->  dst: (N,K) f16
__global__ __launch_bounds__(256) void transpose_f16_kernel(
    const float* __restrict__ src, _Float16* __restrict__ dst, int K, int N)
{
    int id = blockIdx.x * 256 + threadIdx.x;
    if (id >= N * K) return;
    int n = id / K, k = id % K;
    dst[id] = (_Float16)src[(size_t)k * N + n];
}

// biasB[h][n][m] = bias_table[rel_index[n*N+m]][h]   (dense, f32, 3 MB -> L2)
__global__ __launch_bounds__(256) void bias_build_kernel(
    const float* __restrict__ table, const int* __restrict__ rel,
    float* __restrict__ biasB)
{
    int id = blockIdx.x * 256 + threadIdx.x;   // id < N*N = 65536
    int rr = rel[id];
    #pragma unroll
    for (int h = 0; h < HEADS; ++h)
        biasB[h * (NSEQ * NSEQ) + id] = table[rr * HEADS + h];
}

// qkv (B,N,3,H,Dh) f16 -> q(B,H,N,Dh)*scale, k(B,H,N,Dh), vT(B,H,Dh,N)
__global__ __launch_bounds__(256) void repack_kernel(
    const _Float16* __restrict__ qkv,
    _Float16* __restrict__ q, _Float16* __restrict__ kk, _Float16* __restrict__ vT)
{
    long long id = (long long)blockIdx.x * 256 + threadIdx.x;
    if (id >= (long long)MROWS * C3) return;
    int r = (int)(id / C3);       // b*N + n
    int c = (int)(id % C3);
    int b = r >> 8, n = r & 255;
    int which = c / PLANES;
    int cc = c - which * PLANES;
    int h = cc >> 6, d = cc & 63;
    _Float16 val = qkv[id];
    int bh = b * HEADS + h;
    if (which == 0)
        q[((size_t)bh * NSEQ + n) * DH + d] = (_Float16)((float)val * SCALE_Q);
    else if (which == 1)
        kk[((size_t)bh * NSEQ + n) * DH + d] = val;
    else
        vT[((size_t)bh * DH + d) * NSEQ + n] = val;
}

// ===========================================================================
// Fused attention: one workgroup per (b,h). 8 waves.
// Full 256x256 f32 score matrix lives in LDS (266 KB < 320 KB WGP LDS).
// ===========================================================================
#define LDSS 260   // padded row stride (floats) to avoid bank conflicts

__global__ __launch_bounds__(256) void attn_kernel(
    const _Float16* __restrict__ q, const _Float16* __restrict__ kmat,
    const _Float16* __restrict__ vT, const float* __restrict__ biasB,
    _Float16* __restrict__ attn_out)
{
    extern __shared__ float S[];   // [256][LDSS]
    const int bh = blockIdx.x;
    const int b = bh / HEADS, h = bh % HEADS;
    const int lane   = threadIdx.x & 31;
    const int wave   = threadIdx.x >> 5;
    const int lane16 = lane & 15;
    const int laneHi = lane >> 4;

    const _Float16* qb = q    + (size_t)bh * NSEQ * DH;
    const _Float16* kb = kmat + (size_t)bh * NSEQ * DH;
    const _Float16* vb = vT   + (size_t)bh * DH * NSEQ;
    const float*    bp = biasB + (size_t)h * NSEQ * NSEQ;

    // ---- Phase 1: S = (q*scale) @ k^T + bias ------------------------------
    for (int t = wave; t < 256; t += 8) {
        const int ti = t >> 4, tj = t & 15;
        v8f acc = {};
        #pragma unroll
        for (int ks = 0; ks < 2; ++ks) {
            const int k0 = ks * 32;
            const _Float16* ap = qb + (ti * 16 + lane16) * DH + k0 + laneHi * 8;
            HalfVec a;
            a.u[0] = *(const uint4*)(ap);
            a.u[1] = *(const uint4*)(ap + 16);
            const _Float16* bz = kb + (tj * 16 + lane16) * DH + k0 + laneHi * 8;
            HalfVec bb;
            bb.u[0] = *(const uint4*)(bz);
            bb.u[1] = *(const uint4*)(bz + 16);
            acc = __builtin_amdgcn_wmma_f32_16x16x32_f16(
                false, a.h, false, bb.h, (short)0, acc, false, false);
        }
        #pragma unroll
        for (int r = 0; r < 8; ++r) {
            const int row = ti * 16 + laneHi * 8 + r;
            const int col = tj * 16 + lane16;
            S[row * LDSS + col] = acc[r] + bp[row * NSEQ + col];
        }
    }
    __syncthreads();

    // ---- Phase 2: softmax, one thread per row -----------------------------
    {
        float* sr = S + threadIdx.x * LDSS;
        float mx = -3.4e38f;
        for (int m = 0; m < NSEQ; m += 4) {
            float4 f = *(const float4*)(sr + m);
            mx = fmaxf(mx, fmaxf(fmaxf(f.x, f.y), fmaxf(f.z, f.w)));
        }
        float sum = 0.f;
        for (int m = 0; m < NSEQ; m += 4) {
            float4 f = *(float4*)(sr + m);
            f.x = __expf(f.x - mx); f.y = __expf(f.y - mx);
            f.z = __expf(f.z - mx); f.w = __expf(f.w - mx);
            sum += f.x + f.y + f.z + f.w;
            *(float4*)(sr + m) = f;
        }
        const float inv = 1.0f / sum;
        for (int m = 0; m < NSEQ; m += 4) {
            float4 f = *(float4*)(sr + m);
            f.x *= inv; f.y *= inv; f.z *= inv; f.w *= inv;
            *(float4*)(sr + m) = f;
        }
    }
    __syncthreads();

    // ---- Phase 3: O = P @ V  (P from LDS, cvt f32->f16, V^T contiguous) ---
    for (int t = wave; t < 64; t += 8) {
        const int ti = t >> 2, tj = t & 3;
        v8f acc = {};
        for (int ks = 0; ks < 8; ++ks) {
            const int k0 = ks * 32;
            const float* pr = S + (ti * 16 + lane16) * LDSS + k0 + laneHi * 8;
            float4 f0 = *(const float4*)(pr);
            float4 f1 = *(const float4*)(pr + 4);
            float4 f2 = *(const float4*)(pr + 16);
            float4 f3 = *(const float4*)(pr + 20);
            v16h a;
            a[0]  = (_Float16)f0.x; a[1]  = (_Float16)f0.y;
            a[2]  = (_Float16)f0.z; a[3]  = (_Float16)f0.w;
            a[4]  = (_Float16)f1.x; a[5]  = (_Float16)f1.y;
            a[6]  = (_Float16)f1.z; a[7]  = (_Float16)f1.w;
            a[8]  = (_Float16)f2.x; a[9]  = (_Float16)f2.y;
            a[10] = (_Float16)f2.z; a[11] = (_Float16)f2.w;
            a[12] = (_Float16)f3.x; a[13] = (_Float16)f3.y;
            a[14] = (_Float16)f3.z; a[15] = (_Float16)f3.w;
            const _Float16* vp = vb + (tj * 16 + lane16) * NSEQ + k0 + laneHi * 8;
            HalfVec bb;
            bb.u[0] = *(const uint4*)(vp);
            bb.u[1] = *(const uint4*)(vp + 16);
            acc = __builtin_amdgcn_wmma_f32_16x16x32_f16(
                false, a, false, bb.h, (short)0, acc, false, false);
        }
        #pragma unroll
        for (int r = 0; r < 8; ++r) {
            const int n = ti * 16 + laneHi * 8 + r;
            const int d = tj * 16 + lane16;
            attn_out[((size_t)(b * NSEQ + n)) * PLANES + h * DH + d] = (_Float16)acc[r];
        }
    }
}

// ===========================================================================
// Host launcher
// ===========================================================================
extern "C" void kernel_launch(void* const* d_in, const int* in_sizes, int n_in,
                              void* d_out, int out_size, void* d_ws, size_t ws_size,
                              hipStream_t stream)
{
    const float* x          = (const float*)d_in[0];
    const float* qkv_w      = (const float*)d_in[1];
    const float* qkv_b      = (const float*)d_in[2];
    const float* bias_table = (const float*)d_in[3];
    const float* proj_w     = (const float*)d_in[4];
    const float* proj_b     = (const float*)d_in[5];
    const int*   rel_index  = (const int*)d_in[6];
    float*       out        = (float*)d_out;
    char*        ws         = (char*)d_ws;

    // workspace layout (bytes)
    const size_t SZ_X16   = (size_t)MROWS * PLANES * 2;        // 25.2 MB
    const size_t SZ_WQKV  = (size_t)C3 * PLANES * 2;           //  3.5 MB
    const size_t SZ_WPROJ = (size_t)PLANES * PLANES * 2;       //  1.2 MB
    const size_t SZ_BIAS  = (size_t)HEADS * NSEQ * NSEQ * 4;   //  3.1 MB
    const size_t SZ_QKV   = (size_t)MROWS * C3 * 2;            // 75.5 MB
    const size_t SZ_HBUF  = (size_t)MROWS * PLANES * 2;        // 25.2 MB each

    size_t off = 0;
    _Float16* xf16   = (_Float16*)(ws + off); off += SZ_X16;
    _Float16* wqkvT  = (_Float16*)(ws + off); off += SZ_WQKV;
    _Float16* wprojT = (_Float16*)(ws + off); off += SZ_WPROJ;
    float*    biasB  = (float*)   (ws + off); off += SZ_BIAS;
    _Float16* qkvbuf = (_Float16*)(ws + off); off += SZ_QKV;
    _Float16* qbuf   = (_Float16*)(ws + off); off += SZ_HBUF;
    _Float16* kbuf   = (_Float16*)(ws + off); off += SZ_HBUF;
    _Float16* vTbuf  = (_Float16*)(ws + off); off += SZ_HBUF;
    _Float16* attnb  = (_Float16*)(ws + off); off += SZ_HBUF;

    // 1) precision prep
    {
        const int n = MROWS * PLANES;
        f32_to_f16_kernel<<<(n + 255) / 256, 256, 0, stream>>>(x, xf16, n);
    }
    {
        const int n = C3 * PLANES;
        transpose_f16_kernel<<<(n + 255) / 256, 256, 0, stream>>>(qkv_w, wqkvT, PLANES, C3);
    }
    {
        const int n = PLANES * PLANES;
        transpose_f16_kernel<<<(n + 255) / 256, 256, 0, stream>>>(proj_w, wprojT, PLANES, PLANES);
    }
    bias_build_kernel<<<(NSEQ * NSEQ) / 256, 256, 0, stream>>>(bias_table, rel_index, biasB);

    // 2) QKV GEMM (16384 x 2304 x 768), f16 out, async-LDS staged
    gemm_wmma_kernel<1><<<dim3(MROWS / 128, C3 / 128), 128, 0, stream>>>(
        xf16, wqkvT, qkv_b, (void*)qkvbuf, MROWS, C3, PLANES);

    // 3) repack into q (scaled), k, v^T
    {
        const long long n = (long long)MROWS * C3;
        repack_kernel<<<(unsigned)((n + 255) / 256), 256, 0, stream>>>(qkvbuf, qbuf, kbuf, vTbuf);
    }

    // 4) fused attention, one WG per (b,h); 266 KB LDS per WG (<= 320 KB WGP)
    const int SMEM = NSEQ * LDSS * (int)sizeof(float);
    (void)hipFuncSetAttribute((const void*)attn_kernel,
                              hipFuncAttributeMaxDynamicSharedMemorySize, SMEM);
    attn_kernel<<<BATCH * HEADS, 256, SMEM, stream>>>(qbuf, kbuf, vTbuf, biasB, attnb);

    // 5) output projection (16384 x 768 x 768), f32 out + bias
    gemm_wmma_kernel<0><<<dim3(MROWS / 128, PLANES / 128), 128, 0, stream>>>(
        attnb, wprojT, proj_b, (void*)out, MROWS, PLANES, PLANES);
}
